// ScaleDotProductAttention_32195074851425
// MI455X (gfx1250) — matrix operand order repaired and checked by reading, not compiled, not run on hardware
//
#include <hip/hip_runtime.h>
#include <hip/hip_bf16.h>
#include <math.h>

typedef _Float16 v16h __attribute__((ext_vector_type(16)));
typedef _Float16 v8h  __attribute__((ext_vector_type(8)));
typedef _Float16 v4h  __attribute__((ext_vector_type(4)));
typedef float    v8f  __attribute__((ext_vector_type(8)));

#define S_LEN   2048
#define DHEAD   128
#define MTILE   16
#define KCHUNK  256
#define NCHUNKS (S_LEN / KCHUNK)   // 8
#define KSTEPS  (DHEAD / 32)       // 4 WMMA k-steps for Q@K^T
#define PSTEPS  (KCHUNK / 32)      // 8 WMMA k-steps for P@V per chunk
#define SKS     144                // halfs per row for K/V stage (pad for banks, 32B aligned)
#define SQS     144
#define SPS     2064               // halfs per row for f16 probabilities
#define THREADS 256

// LDS carve (bytes)
#define OFF_SS  0                          // f32 scores 16 x 2048      = 131072
#define OFF_SP  131072                     // f16 probs  16 x 2064      =  66048
#define OFF_SK  197120                     // f16 K/V    256 x 144      =  73728
#define OFF_SQ  270848                     // f16 Q      16 x 144       =   4608
#define OFF_RED 275456                     // reduction scratch         =   1152
#define SMEM_TOTAL 276608

// Swap bits 3 and 4 of the within-32 k-index (involution). Storing element k at
// position swz(k) makes the WMMA A-fragment a single contiguous v16h load:
// fragment element e reads position half*16+e, and swz maps that back to the
// required k = (e>>3)*16 + half*8 + (e&7). 4-aligned groups stay contiguous.
__device__ __forceinline__ int swz32(int c) {
  return (c & ~24) | ((c & 8) << 1) | ((c & 16) >> 1);
}

__device__ __forceinline__ v8f wmma_f16(v16h a, v16h b, v8f c) {
  // D = A*B + C, 16x16x32, f16 inputs, f32 accumulate
  return __builtin_amdgcn_wmma_f32_16x16x32_f16(false, a, false, b, (short)0, c,
                                                false, false);
}

__global__ void __launch_bounds__(THREADS, 1)
fa_fused_kernel(const float* __restrict__ q, const float* __restrict__ k,
                const float* __restrict__ v, float* __restrict__ out,
                float* __restrict__ attn)
{
  extern __shared__ char smem[];
  float*    sS  = (float*)(smem + OFF_SS);
  _Float16* sP  = (_Float16*)(smem + OFF_SP);
  _Float16* sKV = (_Float16*)(smem + OFF_SK);
  _Float16* sQ  = (_Float16*)(smem + OFF_SQ);
  float*    red    = (float*)(smem + OFF_RED); // 256 partials
  float*    rowmax = red + 256;                // 16
  float*    rowinv = red + 272;                // 16

  const int t    = threadIdx.x;
  const int lane = t & 31;
  const int wave = t >> 5;     // 0..7
  const int n16  = lane & 15;  // matrix row/col index within half-wave
  const int half = lane >> 4;  // 0/1

  const int bh = blockIdx.x >> 7;              // 0..63 (B*H)
  const int q0 = (blockIdx.x & 127) * MTILE;   // query-tile base

  const size_t head_off = (size_t)bh * S_LEN * DHEAD;
  const float* gQ = q + head_off + (size_t)q0 * DHEAD;
  const float* gK = k + head_off;
  const float* gV = v + head_off;
  float* gOut  = out + head_off + (size_t)q0 * DHEAD;
  float* gAttn = attn + ((size_t)bh * S_LEN + q0) * S_LEN;

  const float scale = 0.08838834764831845f;    // 1/sqrt(128)

  // ---- Stage Q tile (16x128 f32 -> f16 LDS, row-major, unswizzled), once ----
  {
    const float4* gq4 = (const float4*)gQ;
    #pragma unroll
    for (int i = 0; i < (MTILE * DHEAD) / (4 * THREADS); ++i) { // 2
      int idx4 = i * THREADS + t;
      float4 x = gq4[idx4];
      int e = idx4 << 2;
      int r = e >> 7, c = e & 127;
      int cs = c;
      v4h h = {(_Float16)x.x, (_Float16)x.y, (_Float16)x.z, (_Float16)x.w};
      *(v4h*)(sQ + r * SQS + cs) = h;
    }
  }
  __syncthreads();

  // ---- Hoist A fragments (Q): one aligned v16h load per k-step ----
  v16h aQ[KSTEPS];
  #pragma unroll
  for (int ks = 0; ks < KSTEPS; ++ks)
    aQ[ks] = *(const v16h*)(sQ + n16 * SQS + ks * 32 + half * 16);

  // ---- Phase 1: scores = scale * Q @ K^T into LDS (16 x 2048 f32) ----
  for (int c = 0; c < NCHUNKS; ++c) {
    __syncthreads();
    { // stage K chunk: 256 keys x 128 dims, f32 -> f16, coalesced
      const float4* gk4 = (const float4*)(gK + (size_t)c * KCHUNK * DHEAD);
      #pragma unroll
      for (int i = 0; i < (KCHUNK * DHEAD) / (4 * THREADS); ++i) { // 32
        int idx4 = i * THREADS + t;
        float4 x = gk4[idx4];
        int e = idx4 << 2;
        int r = e >> 7, cc = e & 127;
        v4h h = {(_Float16)x.x, (_Float16)x.y, (_Float16)x.z, (_Float16)x.w};
        *(v4h*)(sKV + r * SKS + cc) = h;
      }
    }
    // prefetch next K chunk into cache while this chunk's WMMAs run
    if (c + 1 < NCHUNKS) {
      const char* nb = (const char*)(gK + (size_t)(c + 1) * KCHUNK * DHEAD);
      #pragma unroll
      for (int j = 0; j < 4; ++j)
        __builtin_prefetch(nb + (size_t)t * 512 + j * 128, 0, 1);
    }
    __syncthreads();

    // Preload ALL B fragments for this chunk (distinct registers -> the 16
    // ds_load_b128 issue back-to-back and overlap the WMMA sequence), then
    // run WMMAs alternating between the two independent accumulator chains.
    v16h bfr[2][KSTEPS];
    #pragma unroll
    for (int tile = 0; tile < 2; ++tile) {
      int keyrow = wave * 32 + tile * 16 + n16;   // B col n = key index
      #pragma unroll
      for (int ks = 0; ks < KSTEPS; ++ks)
        bfr[tile][ks] = *(const v16h*)(sKV + keyrow * SKS + ks * 32 + half * 16);
    }
    v8f acc0 = {}, acc1 = {};
    #pragma unroll
    for (int ks = 0; ks < KSTEPS; ++ks) {
      acc0 = wmma_f16(aQ[ks], bfr[0][ks], acc0);
      acc1 = wmma_f16(aQ[ks], bfr[1][ks], acc1);
    }
    int col0 = c * KCHUNK + wave * 32 + n16;
    #pragma unroll
    for (int r = 0; r < 8; ++r) {                 // C: M = half*8 + r, N = n16
      sS[(half * 8 + r) * S_LEN + col0]      = acc0[r] * scale;
      sS[(half * 8 + r) * S_LEN + col0 + 16] = acc1[r] * scale;
    }
  }
  __syncthreads();

  // ---- Phase 2: exact row softmax over LDS strip ----
  { // partial row max (16 threads per row, stride-16 conflict-free sweep)
    int row = t >> 4, seg = t & 15;
    const float* srow = sS + row * S_LEN;
    float mx = -3.4e38f;
    for (int j = 0; j < S_LEN / 16; ++j) mx = fmaxf(mx, srow[seg + (j << 4)]);
    red[t] = mx;
  }
  __syncthreads();
  if (t < MTILE) {
    float mx = red[t * 16];
    #pragma unroll
    for (int j = 1; j < 16; ++j) mx = fmaxf(mx, red[t * 16 + j]);
    rowmax[t] = mx;
  }
  __syncthreads();
  { // exp(s - max) in place + partial sums
    int row = t >> 4, seg = t & 15;
    float m = rowmax[row];
    float* srow = sS + row * S_LEN;
    float s = 0.f;
    for (int j = 0; j < S_LEN / 16; ++j) {
      int idx = seg + (j << 4);
      float e = __expf(srow[idx] - m);
      srow[idx] = e;
      s += e;
    }
    red[t] = s;
  }
  __syncthreads();
  if (t < MTILE) {
    float s = 0.f;
    #pragma unroll
    for (int j = 0; j < 16; ++j) s += red[t * 16 + j];
    rowinv[t] = 1.0f / s;
  }
  __syncthreads();
  { // normalize: f32 attention to HBM (coalesced) + row-major f16 copy to LDS
    #pragma unroll
    for (int i = 0; i < (MTILE * S_LEN) / (4 * THREADS); ++i) { // 32
      int idx4 = i * THREADS + t;
      int e = idx4 << 2;
      int row = e >> 11, col = e & 2047;
      float inv = rowinv[row];
      float4 x = *(const float4*)(sS + row * S_LEN + col);
      float4 p = make_float4(x.x * inv, x.y * inv, x.z * inv, x.w * inv);
      *(float4*)(gAttn + (size_t)row * S_LEN + col) = p;
      int cs = col;
      v4h h = {(_Float16)p.x, (_Float16)p.y, (_Float16)p.z, (_Float16)p.w};
      *(v4h*)(sP + row * SPS + cs) = h;
    }
  }

  // ---- Phase 3: out = P @ V. Wave w owns dims [w*16, w*16+16) ----
  v8f accO0 = {}, accO1 = {};
  for (int c = 0; c < NCHUNKS; ++c) {
    __syncthreads();
    { // stage V chunk f32 -> f16 (reuse K stage region)
      const float4* gv4 = (const float4*)(gV + (size_t)c * KCHUNK * DHEAD);
      #pragma unroll
      for (int i = 0; i < (KCHUNK * DHEAD) / (4 * THREADS); ++i) { // 32
        int idx4 = i * THREADS + t;
        float4 x = gv4[idx4];
        int e = idx4 << 2;
        int r = e >> 7, cc = e & 127;
        v4h h = {(_Float16)x.x, (_Float16)x.y, (_Float16)x.z, (_Float16)x.w};
        *(v4h*)(sKV + r * SKS + cc) = h;
      }
    }
    // prefetch next V chunk
    if (c + 1 < NCHUNKS) {
      const char* nb = (const char*)(gV + (size_t)(c + 1) * KCHUNK * DHEAD);
      #pragma unroll
      for (int j = 0; j < 4; ++j)
        __builtin_prefetch(nb + (size_t)t * 512 + j * 128, 0, 1);
    }
    __syncthreads();

    // Preload all A fragments (vector loads) and all gathered B fragments for
    // this chunk, then run the 8 WMMAs alternating across two accumulators.
    v16h aP[PSTEPS];
    #pragma unroll
    for (int ks = 0; ks < PSTEPS; ++ks)
      aP[ks] = *(const v16h*)(sP + n16 * SPS + c * KCHUNK + ks * 32 + half * 16);
    v16h bV[PSTEPS];
    #pragma unroll
    for (int ks = 0; ks < PSTEPS; ++ks) {
      // B element e -> key k = ks*32 + half*16 + e, dim = wave*16 + n16
      const _Float16* vcol = sKV + (ks * 32 + half * 16) * SKS + wave * 16 + n16;
      #pragma unroll
      for (int e = 0; e < 16; ++e) bV[ks][e] = vcol[e * SKS];
    }
    #pragma unroll
    for (int ks = 0; ks < PSTEPS; ks += 2) {
      accO0 = wmma_f16(aP[ks],     bV[ks],     accO0);
      accO1 = wmma_f16(aP[ks + 1], bV[ks + 1], accO1);
    }
  }
  // write out tile: row = half*8 + r, dim = wave*16 + n16
  #pragma unroll
  for (int r = 0; r < 8; ++r) {
    gOut[(half * 8 + r) * DHEAD + wave * 16 + n16] = accO0[r] + accO1[r];
  }
}

extern "C" void kernel_launch(void* const* d_in, const int* in_sizes, int n_in,
                              void* d_out, int out_size, void* d_ws, size_t ws_size,
                              hipStream_t stream) {
  (void)in_sizes; (void)n_in; (void)d_ws; (void)ws_size; (void)out_size;
  const float* q = (const float*)d_in[0];
  const float* k = (const float*)d_in[1];
  const float* v = (const float*)d_in[2];
  float* out  = (float*)d_out;
  float* attn = out + (size_t)4 * 16 * S_LEN * DHEAD;  // out first, then attention

  hipFuncSetAttribute((const void*)fa_fused_kernel,
                      hipFuncAttributeMaxDynamicSharedMemorySize, SMEM_TOTAL);

  dim3 grid(64 * (S_LEN / MTILE));   // 8192 workgroups; same-head blocks adjacent (L2 reuse)
  dim3 block(THREADS);
  fa_fused_kernel<<<grid, block, SMEM_TOTAL, stream>>>(q, k, v, out, attn);
}
